// Layer_84559316124234
// MI455X (gfx1250) — matrix operand order repaired and checked by reading, not compiled, not run on hardware
//
#include <hip/hip_runtime.h>
#include <hip/hip_bf16.h>

// Problem constants (from reference)
#define B_   128
#define P_   4
#define N_   128
#define T_   2048
#define EPS_ 1e-5f

typedef __attribute__((ext_vector_type(16))) __bf16 v16bf;
typedef __attribute__((ext_vector_type(8)))  __bf16 v8bf;
typedef __attribute__((ext_vector_type(8)))  float  v8f;

__device__ __forceinline__ float sigmoidf_(float z) {
    return 1.0f / (1.0f + __expf(-z));
}

// ---------------------------------------------------------------------------
// Kernel 1: layernorm over N (axis=2) of xs (B, 8, N, T), fused with a
// transpose into scratch layout xn (T, B, 8, N) so the scan reads coalesced
// per-timestep slabs. One WG per (b,pp, t-block of 128).
// ---------------------------------------------------------------------------
__global__ __launch_bounds__(256) void norm_transpose_kernel(
    const float* __restrict__ xs,
    const float* __restrict__ gamma,   // (8, N)
    const float* __restrict__ beta,    // (8, N)
    float* __restrict__ xn)            // (T, B, 8, N)
{
    extern __shared__ char smem[];
    float* tile = (float*)smem;          // 128 x 129 (padded: bank-conflict free)
    float* mcol = tile + 128 * 129;      // 128 means
    float* icol = mcol + 128;            // 128 inv-std

    const int TB  = 128;
    const int tb  = blockIdx.x & 15;     // T_/TB = 16 t-blocks
    const int bp  = blockIdx.x >> 4;     // 0..1023 == b*8+pp
    const int b   = bp >> 3;
    const int pp  = bp & 7;
    const int t0  = tb * TB;
    const int tid = threadIdx.x;

    for (int i = 0; i < 64; ++i) {
        int f  = tid + 256 * i;
        int tt = f & 127;
        int n  = f >> 7;
        tile[n * 129 + tt] = xs[((size_t)bp * 128 + n) * 2048 + t0 + tt];
    }
    __syncthreads();

    if (tid < 128) {
        float s = 0.f, ss = 0.f;
        for (int n = 0; n < 128; ++n) {
            float v = tile[n * 129 + tid];
            s += v; ss += v * v;
        }
        float m = s * (1.0f / 128.0f);
        float var = ss * (1.0f / 128.0f) - m * m;
        mcol[tid] = m;
        icol[tid] = rsqrtf(var + EPS_);
    }
    __syncthreads();

    for (int i = 0; i < 64; ++i) {
        int f  = tid + 256 * i;
        int n  = f & 127;
        int tt = f >> 7;
        float v = (tile[n * 129 + tt] - mcol[tt]) * icol[tt] * gamma[pp * 128 + n]
                  + beta[pp * 128 + n];
        xn[(((size_t)(t0 + tt) * 128 + b) * 8 + pp) * 128 + n] = v;
    }
}

// ---------------------------------------------------------------------------
// Kernel 2: persistent recurrent scan. 16 WGs = 8 batch tiles (M=16) x 2
// independent integrators. bf16 weights + bf16 state shadows in LDS laid out
// so every WMMA fragment is 2x ds_load_b128; input slabs double-buffered in
// LDS via async global->LDS loads (ASYNCcnt).
// ---------------------------------------------------------------------------
__global__ __launch_bounds__(256) void scan_kernel(
    const float* __restrict__ xn,       // (T, B, 8, N) normalized
    const float* __restrict__ init_h,   // (B, 2N)
    const float* __restrict__ aflxG, const float* __restrict__ aflxB,
    const float* __restrict__ aextG, const float* __restrict__ aextB,
    const float* __restrict__ fxW, const float* __restrict__ fxw, const float* __restrict__ fxb,
    const float* __restrict__ fhW, const float* __restrict__ fhw, const float* __restrict__ fhb,
    const float* __restrict__ exW, const float* __restrict__ exw, const float* __restrict__ exb,
    const float* __restrict__ ehW, const float* __restrict__ ehw, const float* __restrict__ ehb,
    float* __restrict__ out)            // (B, 2N, T)
{
    const int integ = blockIdx.x & 1;      // 0 = flx, 1 = ext
    const int tileb = blockIdx.x >> 1;     // 0..7
    const int b0    = tileb * 16;

    const float* aG_g   = integ ? aextG : aflxG;
    const float* aB_g   = integ ? aextB : aflxB;
    const float* Wx_g   = integ ? exW : fxW;   // (128,128)  x2h
    const float* lnw2_g = integ ? exw : fxw;
    const float* lnb2_g = integ ? exb : fxb;
    const float* Wh_g   = integ ? ehW : fhW;   // (256,128)  h2h
    const float* lnw1_g = integ ? ehw : fhw;
    const float* lnb1_g = integ ? ehb : fhb;

    // ---- LDS carve-up (bf16 rows stride 136 = 272B: 16B-aligned b128) ----
    extern __shared__ char smem[];
    __bf16* Wh  = (__bf16*)(smem);                 // 256 x 136          (69632)
    __bf16* Wx  = (__bf16*)(smem + 69632);         // 128 x 136          (34816)
    __bf16* hxb = (__bf16*)(smem + 104448);        // 16 x 136 bf16 hx    (4352)
    __bf16* xvb = (__bf16*)(smem + 108800);        // 16 x 136 bf16 x     (4352)
    float*  hx  = (float*)(smem + 113152);         // 16 x 132 f32 hx     (8448)
    float*  inx = (float*)(smem + 121600);         // 16 x 132 gated in   (8448)
    float*  hh  = (float*)(smem + 130048);         // 16 x 264 h2h out   (16896)
    float*  lnw1 = (float*)(smem + 146944);        // 256
    float*  lnb1 = lnw1 + 256;                     // 256
    float*  lnw2 = lnb1 + 256;                     // 128
    float*  lnb2 = lnw2 + 128;                     // 128
    float*  aG   = lnb2 + 128;                     // 512
    float*  aB   = aG + 512;                       // 512
    float*  xbuf = (float*)(smem + 154112);        // 2 x 16 x 512 dbl buf (65536)

    const int tid   = threadIdx.x;
    const int lane  = tid & 31;
    const int wave  = tid >> 5;
    const int mA    = lane & 15;
    const int halfA = lane >> 4;

    // ---- one-time setup ----
    for (int i = 0; i < 128; ++i) {                   // h2h weights 256x128
        int f = tid + 256 * i;
        Wh[(f >> 7) * 136 + (f & 127)] = (__bf16)Wh_g[f];
    }
    for (int i = 0; i < 64; ++i) {                    // x2h weights 128x128
        int f = tid + 256 * i;
        Wx[(f >> 7) * 136 + (f & 127)] = (__bf16)Wx_g[f];
    }
    for (int i = 0; i < 8; ++i) {                     // hx tile 16x128
        int f = tid + 256 * i;
        int rr = f >> 7, n = f & 127;
        float h0 = init_h[(size_t)(b0 + rr) * 256 + integ * 128 + n];
        hx[rr * 132 + n]  = h0;
        hxb[rr * 136 + n] = (__bf16)h0;
    }
    { lnw1[tid] = lnw1_g[tid]; lnb1[tid] = lnb1_g[tid]; }
    if (tid < 128) { lnw2[tid] = lnw2_g[tid]; lnb2[tid] = lnb2_g[tid]; }
    for (int i = 0; i < 2; ++i) {
        int f = tid + 256 * i;
        aG[f] = aG_g[f]; aB[f] = aB_g[f];
    }

    // Prologue: async-stage slab for t=0 into buffer 0 (16B per lane-op).
    {
        float* dst = xbuf;
        const float* src = xn + ((size_t)0 * 128 + b0) * 1024 + integ * 512;
#pragma unroll
        for (int i = 0; i < 8; ++i) {
            int c   = tid + 256 * i;        // 2048 x 16B chunks
            int rr  = c >> 7;
            int off = (c & 127) * 4;        // float index in row
            unsigned ldsa  = (unsigned)(size_t)(dst + rr * 512 + off);
            unsigned long long ga = (unsigned long long)(size_t)(src + (size_t)rr * 1024 + off);
            asm volatile("global_load_async_to_lds_b128 %0, %1, off"
                         :: "v"(ldsa), "v"(ga) : "memory");
        }
    }
    __syncthreads();

    v16bf afr[4];

    for (int t = 0; t < T_; ++t) {
        const int cur = t & 1;
        const int nxt = cur ^ 1;

        // Wait for this step's staged slab (issued last iteration / prologue),
        // then make it visible to all waves.
        asm volatile("s_wait_asynccnt 0x0" ::: "memory");
        __syncthreads();

        // Kick off async staging of next step's slab into the other buffer.
        if (t + 1 < T_) {
            float* dst = xbuf + nxt * 8192;
            const float* src = xn + ((size_t)(t + 1) * 128 + b0) * 1024 + integ * 512;
#pragma unroll
            for (int i = 0; i < 8; ++i) {
                int c   = tid + 256 * i;
                int rr  = c >> 7;
                int off = (c & 127) * 4;
                unsigned ldsa  = (unsigned)(size_t)(dst + rr * 512 + off);
                unsigned long long ga = (unsigned long long)(size_t)(src + (size_t)rr * 1024 + off);
                asm volatile("global_load_async_to_lds_b128 %0, %1, off"
                             :: "v"(ldsa), "v"(ga) : "memory");
            }
        }

        // ---- Phase A: in = relu(sum_p sigmoid(h*aG+aB)*x), x from LDS ----
        const float* xcur = xbuf + cur * 8192;
        for (int i = 0; i < 8; ++i) {
            int f  = tid + 256 * i;
            int rr = f >> 7, n = f & 127;
            float h = hx[rr * 132 + n];
            float s = 0.f;
#pragma unroll
            for (int p = 0; p < 4; ++p) {
                float g = sigmoidf_(h * aG[p * 128 + n] + aB[p * 128 + n]);
                s += g * xcur[rr * 512 + p * 128 + n];
            }
            inx[rr * 132 + n] = fmaxf(s, 0.f);
        }

        // ---- h2h GEMM: (16x128)@(128x256), bf16 WMMA --------------------
        // A frag chunk c: VGPRs 0-3 hold K=half*8..+7, VGPRs 4-7 K=16+half*8..+7
        // -> two aligned 16B LDS loads from the bf16 shadow.
#pragma unroll
        for (int c = 0; c < 4; ++c) {
            const __bf16* abase = hxb + mA * 136 + c * 32;
            v8bf lo = *(const v8bf*)(abase + halfA * 8);
            v8bf hi = *(const v8bf*)(abase + 16 + halfA * 8);
            afr[c] = __builtin_shufflevector(lo, hi, 0,1,2,3,4,5,6,7,8,9,10,11,12,13,14,15);
        }
#pragma unroll
        for (int u = 0; u < 2; ++u) {
            int jb = (wave * 2 + u) * 16;
            int jc = jb + mA;
            v8f acc = {0.f, 0.f, 0.f, 0.f, 0.f, 0.f, 0.f, 0.f};
#pragma unroll
            for (int c = 0; c < 4; ++c) {
                // B frag chunk c: VGPRs 0-3 hold K=half*8..+7, VGPRs 4-7
                // K=16+half*8..+7 = two aligned 16B loads.
                const __bf16* bbase = Wh + jc * 136 + c * 32;
                v8bf b0 = *(const v8bf*)(bbase + halfA * 8);
                v8bf b1 = *(const v8bf*)(bbase + 16 + halfA * 8);
                v16bf bfr = __builtin_shufflevector(b0, b1, 0,1,2,3,4,5,6,7,8,9,10,11,12,13,14,15);
                acc = __builtin_amdgcn_wmma_f32_16x16x32_bf16(
                          false, afr[c], false, bfr, (short)0, acc, false, false);
            }
#pragma unroll
            for (int r = 0; r < 8; ++r)
                hh[(halfA * 8 + r) * 264 + jb + mA] = acc[r];
        }
        __syncthreads();   // B1: inx + hh ready

        // ---- LN(256) -> h_i,h_g; x = relu(in + h_i); keep x,h_g in regs --
        float xreg[2][4], hgreg[2][4];
#pragma unroll
        for (int rw = 0; rw < 2; ++rw) {
            int rr = wave * 2 + rw;
            float s = 0.f, ss = 0.f;
#pragma unroll
            for (int q = 0; q < 8; ++q) {
                float v = hh[rr * 264 + lane + 32 * q];
                s += v; ss += v * v;
            }
#pragma unroll
            for (int off = 16; off > 0; off >>= 1) {
                s  += __shfl_xor(s,  off, 32);
                ss += __shfl_xor(ss, off, 32);
            }
            float m    = s * (1.f / 256.f);
            float istd = rsqrtf(ss * (1.f / 256.f) - m * m + EPS_);
#pragma unroll
            for (int q = 0; q < 4; ++q) {
                int n = lane + 32 * q;
                float hi  = (hh[rr * 264 + n]       - m) * istd * lnw1[n]       + lnb1[n];
                float hgv = (hh[rr * 264 + 128 + n] - m) * istd * lnw1[128 + n] + lnb1[128 + n];
                float xr  = fmaxf(inx[rr * 132 + n] + hi, 0.f);
                xreg[rw][q]  = xr;
                hgreg[rw][q] = hgv;
                xvb[rr * 136 + n] = (__bf16)xr;
            }
        }
        __syncthreads();   // B2: xvb ready, hh free for reuse

        // ---- x2h GEMM: (16x128)@(128x128), bf16 WMMA --------------------
#pragma unroll
        for (int c = 0; c < 4; ++c) {
            const __bf16* abase = xvb + mA * 136 + c * 32;
            v8bf lo = *(const v8bf*)(abase + halfA * 8);
            v8bf hi = *(const v8bf*)(abase + 16 + halfA * 8);
            afr[c] = __builtin_shufflevector(lo, hi, 0,1,2,3,4,5,6,7,8,9,10,11,12,13,14,15);
        }
        {
            int jb = wave * 16;
            int jc = jb + mA;
            v8f acc = {0.f, 0.f, 0.f, 0.f, 0.f, 0.f, 0.f, 0.f};
#pragma unroll
            for (int c = 0; c < 4; ++c) {
                const __bf16* bbase = Wx + jc * 136 + c * 32;
                v8bf b0 = *(const v8bf*)(bbase + halfA * 8);
                v8bf b1 = *(const v8bf*)(bbase + 16 + halfA * 8);
                v16bf bfr = __builtin_shufflevector(b0, b1, 0,1,2,3,4,5,6,7,8,9,10,11,12,13,14,15);
                acc = __builtin_amdgcn_wmma_f32_16x16x32_bf16(
                          false, afr[c], false, bfr, (short)0, acc, false, false);
            }
#pragma unroll
            for (int r = 0; r < 8; ++r)
                hh[(halfA * 8 + r) * 132 + jb + mA] = acc[r];  // out2, stride 132
        }
        __syncthreads();   // B3: out2 ready

        // ---- LN(128) -> g; h = (1-g)*hx + g*x; emit + update state ------
#pragma unroll
        for (int rw = 0; rw < 2; ++rw) {
            int rr = wave * 2 + rw;
            float s = 0.f, ss = 0.f;
#pragma unroll
            for (int q = 0; q < 4; ++q) {
                float v = hh[rr * 132 + lane + 32 * q];
                s += v; ss += v * v;
            }
#pragma unroll
            for (int off = 16; off > 0; off >>= 1) {
                s  += __shfl_xor(s,  off, 32);
                ss += __shfl_xor(ss, off, 32);
            }
            float m    = s * (1.f / 128.f);
            float istd = rsqrtf(ss * (1.f / 128.f) - m * m + EPS_);
#pragma unroll
            for (int q = 0; q < 4; ++q) {
                int n = lane + 32 * q;
                float l2   = (hh[rr * 132 + n] - m) * istd * lnw2[n] + lnb2[n];
                float g    = sigmoidf_(l2 + hgreg[rw][q]);
                float xr   = xreg[rw][q];
                float hold = hx[rr * 132 + n];
                float hnew = (1.f - g) * hold + g * xr;
                hx[rr * 132 + n]  = hnew;
                hxb[rr * 136 + n] = (__bf16)hnew;
                out[((size_t)(b0 + rr) * 256 + integ * 128 + n) * 2048 + t] = hnew;
            }
        }
        __syncthreads();   // B4: state updated for next step
    }
}

extern "C" void kernel_launch(void* const* d_in, const int* in_sizes, int n_in,
                              void* d_out, int out_size, void* d_ws, size_t ws_size,
                              hipStream_t stream) {
    const float* xs     = (const float*)d_in[0];
    const float* init_h = (const float*)d_in[1];
    const float* ng     = (const float*)d_in[2];
    const float* nb     = (const float*)d_in[3];
    const float* aflxG  = (const float*)d_in[4];
    const float* aflxB  = (const float*)d_in[5];
    const float* aextG  = (const float*)d_in[6];
    const float* aextB  = (const float*)d_in[7];
    const float* fxW    = (const float*)d_in[8];
    const float* fxw    = (const float*)d_in[9];
    const float* fxb    = (const float*)d_in[10];
    const float* fhW    = (const float*)d_in[11];
    const float* fhw    = (const float*)d_in[12];
    const float* fhb    = (const float*)d_in[13];
    const float* exW    = (const float*)d_in[14];
    const float* exw    = (const float*)d_in[15];
    const float* exb    = (const float*)d_in[16];
    const float* ehW    = (const float*)d_in[17];
    const float* ehw    = (const float*)d_in[18];
    const float* ehb    = (const float*)d_in[19];
    float* out = (float*)d_out;
    float* xn  = (float*)d_ws;   // (T, B, 8, N) = 1 GiB scratch

    // Pass 1: layernorm over N + transpose to T-outermost.
    size_t smem1 = (size_t)(128 * 129 + 256) * sizeof(float);
    norm_transpose_kernel<<<16384, 256, smem1, stream>>>(xs, ng, nb, xn);

    // Pass 2: persistent recurrent scan, 16 independent chains.
    size_t smem2 = 219648;
    scan_kernel<<<16, 256, smem2, stream>>>(xn, init_h,
        aflxG, aflxB, aextG, aextB,
        fxW, fxw, fxb, fhW, fhw, fhb,
        exW, exw, exb, ehW, ehw, ehb, out);
}
